// DGCNN_CLS_Encoder_7997229105804
// MI455X (gfx1250) — matrix operand order repaired and checked by reading, not compiled, not run on hardware
//
#include <hip/hip_runtime.h>
#include <cstdint>

#define BB 32
#define NN 1024
#define KNNK 20
#define BNEPS 1e-5f
#define SLOPE 0.2f

typedef __attribute__((ext_vector_type(16))) __bf16       v16bf;
typedef __attribute__((ext_vector_type(8)))  float        v8f;
typedef __attribute__((ext_vector_type(8)))  unsigned int v8u;

__device__ __forceinline__ unsigned short f2bf(float f) {
    unsigned u = __builtin_bit_cast(unsigned, f);
    unsigned r = u + 0x7FFFu + ((u >> 16) & 1u);   // round-to-nearest-even
    return (unsigned short)(r >> 16);
}
__device__ __forceinline__ float bf2f(unsigned short s) {
    unsigned u = ((unsigned)s) << 16;
    return __builtin_bit_cast(float, u);
}

// CDNA5 async gather: memory -> LDS, 16B per lane, tracked by ASYNCcnt.
__device__ __forceinline__ void async_gather_b128(const void* gsrc, void* lds_dst) {
    unsigned ldsoff = (unsigned)(uintptr_t)lds_dst;      // LDS aperture: low 32 bits = LDS offset
    unsigned long long ga = (unsigned long long)(uintptr_t)gsrc;
    asm volatile("global_load_async_to_lds_b128 %0, %1, off"
                 :: "v"(ldsoff), "v"(ga) : "memory");
}
__device__ __forceinline__ void wait_async0() {
    asm volatile("s_wait_asynccnt 0x0" ::: "memory");
}

// ---------------------------------------------------------------------------
// Packing: f32 -> zero-padded bf16 rows
// ---------------------------------------------------------------------------
__global__ void pack_rows(const float* __restrict__ src, unsigned short* __restrict__ dst,
                          int rows, int C, int CP) {
    int i = blockIdx.x * blockDim.x + threadIdx.x;
    if (i >= rows * CP) return;
    int r = i / CP, c = i % CP;
    dst[i] = (c < C) ? f2bf(src[(size_t)r * C + c]) : (unsigned short)0;
}

// W (O,2C) -> (O,2CP): first half [0,C) -> [0,C), second half [C,2C) -> [CP,CP+C)
__global__ void pack_w_split(const float* __restrict__ W, unsigned short* __restrict__ dst,
                             int O, int C, int CP) {
    int i = blockIdx.x * blockDim.x + threadIdx.x;
    int TW = 2 * CP;
    if (i >= O * TW) return;
    int o = i / TW, c = i % TW;
    float v = 0.f;
    if (c < C)                      v = W[(size_t)o * 2 * C + c];
    else if (c >= CP && c < CP + C) v = W[(size_t)o * 2 * C + C + (c - CP)];
    dst[i] = f2bf(v);
}

// ---------------------------------------------------------------------------
// KNN: one wave per point; f32 distances in LDS, 20x iterative argmax
// pd = -||xi - xj||^2, top-k largest (self included, matching reference)
// ---------------------------------------------------------------------------
__global__ __launch_bounds__(256) void knn_kernel(const float* __restrict__ x, int C, int ldx,
                                                  int* __restrict__ idx) {
    __shared__ float dist[8][NN];
    __shared__ __align__(16) float xi[8][128];
    int wave = threadIdx.x >> 5, lane = threadIdx.x & 31;
    int gi = blockIdx.x * 8 + wave;
    int b = gi / NN, n = gi % NN;

    const float* xrow = x + (size_t)(b * NN + n) * ldx;
    for (int c = lane; c < C; c += 32) xi[wave][c] = xrow[c];
    __builtin_amdgcn_wave_barrier();

    if (C == 3) {
        float x0 = xi[wave][0], x1 = xi[wave][1], x2 = xi[wave][2];
        for (int j = lane; j < NN; j += 32) {
            const float* xj = x + (size_t)(b * NN + j) * ldx;
            float t0 = x0 - xj[0], t1 = x1 - xj[1], t2 = x2 - xj[2];
            dist[wave][j] = -(t0 * t0 + t1 * t1 + t2 * t2);
        }
    } else {
        for (int j = lane; j < NN; j += 32) {
            const float* xj = x + (size_t)(b * NN + j) * ldx;
            float d = 0.f;
            #pragma unroll 4
            for (int c = 0; c < C; c += 4) {
                float4 a = *(const float4*)&xi[wave][c];
                float4 q = *(const float4*)&xj[c];
                float t0 = a.x - q.x, t1 = a.y - q.y, t2 = a.z - q.z, t3 = a.w - q.w;
                d -= t0 * t0 + t1 * t1 + t2 * t2 + t3 * t3;
            }
            dist[wave][j] = d;
        }
    }
    __builtin_amdgcn_wave_barrier();

    int* idr = idx + (size_t)(b * NN + n) * KNNK;
    for (int kk = 0; kk < KNNK; ++kk) {
        float bv = -__builtin_inff(); int bi = 0;
        for (int j = lane; j < NN; j += 32) {
            float v = dist[wave][j];
            if (v > bv || (v == bv && j < bi)) { bv = v; bi = j; }
        }
        #pragma unroll
        for (int off = 16; off; off >>= 1) {
            float ov = __shfl_xor(bv, off);
            int   oi = __shfl_xor(bi, off);
            if (ov > bv || (ov == bv && oi < bi)) { bv = ov; bi = oi; }
        }
        if (lane == 0) { idr[kk] = bi; dist[wave][bi] = -__builtin_inff(); }
        __builtin_amdgcn_wave_barrier();
    }
}

// ---------------------------------------------------------------------------
// Edge-conv GEMM via WMMA bf16. One block = (b, 8 points) -> 160 edge rows.
// Neighbor rows are gathered memory->LDS with GLOBAL_LOAD_ASYNC_TO_LDS_B128
// (ASYNCcnt), edge matrix [nbr-center | center] is built in place, then
// multiplied by W^T tiles. Fused epilogue: max/min over k + BN-stat atomics.
// ---------------------------------------------------------------------------
template <int CP>
__global__ __launch_bounds__(128) void edge_gemm(
    const unsigned short* __restrict__ xb, int ldx, const int* __restrict__ idx,
    const unsigned short* __restrict__ Wb, int O,
    float* __restrict__ outmax, float* __restrict__ outmin, float* __restrict__ stats) {
    constexpr int TWOC = 2 * CP;
    constexpr int ROWS = 8 * KNNK;     // 160 edge rows (10 M-tiles of 16)
    constexpr int C8   = CP / 8;       // 16-byte chunks per input row
    __shared__ __align__(16) unsigned short E[ROWS * TWOC];
    __shared__ __align__(16) unsigned short CTR[8 * CP];
    __shared__ float stage[4][ROWS * 16];
    __shared__ int sidx[ROWS];

    int tid = threadIdx.x, wave = tid >> 5, lane = tid & 31;
    int b = blockIdx.x / (NN / 8);
    int n0 = (blockIdx.x % (NN / 8)) * 8;

    for (int r = tid; r < ROWS; r += 128)
        sidx[r] = idx[(size_t)(b * NN + n0 + r / KNNK) * KNNK + (r % KNNK)];
    __syncthreads();

    // Stage the 8 center rows (vectorized 16B loads)
    for (int t = tid; t < 8 * C8; t += 128) {
        int n = t / C8, cc = (t % C8) * 8;
        *(uint4*)&CTR[n * CP + cc] =
            *(const uint4*)(xb + (size_t)(b * NN + n0 + n) * ldx + cc);
    }
    // Async-gather the 160 neighbor rows straight into the low half of E rows
    for (int t = tid; t < ROWS * C8; t += 128) {
        int r = t / C8, cc = (t % C8) * 8;
        int j = sidx[r];
        async_gather_b128(xb + (size_t)(b * NN + j) * ldx + cc, &E[r * TWOC + cc]);
    }
    wait_async0();
    __syncthreads();

    // In-place edge build: E[r] = [nbr - center | center]
    for (int e = tid; e < ROWS * CP; e += 128) {
        int r = e / CP, c = e % CP;
        unsigned short ctr = CTR[(r / KNNK) * CP + c];
        unsigned short nbr = E[r * TWOC + c];
        E[r * TWOC + c]      = f2bf(bf2f(nbr) - bf2f(ctr));
        E[r * TWOC + CP + c] = ctr;
    }
    __syncthreads();

    const int otpw = O / 64;           // 16-col O tiles per wave
    for (int oi = 0; oi < otpw; ++oi) {
        int obase = (wave * otpw + oi) * 16;
        __builtin_prefetch(&Wb[(size_t)(obase + (lane & 15)) * TWOC], 0, 1);

        for (int mt = 0; mt < ROWS / 16; ++mt) {
            v8f acc = {0.f, 0.f, 0.f, 0.f, 0.f, 0.f, 0.f, 0.f};
            int m = lane & 15, hi = lane >> 4;
            const unsigned* Erow = (const unsigned*)&E[(mt * 16 + m) * TWOC];
            const unsigned* Wrow = (const unsigned*)&Wb[(size_t)(obase + m) * TWOC];
            #pragma unroll
            for (int ks = 0; ks < TWOC / 32; ++ks) {
                // A fragment (ISA 16-bit A 16x32 layout): packed k-pairs
                int kb = ks * 16 + hi * 4;
                v8u au;
                au[0] = Erow[kb + 0]; au[1] = Erow[kb + 1];
                au[2] = Erow[kb + 2]; au[3] = Erow[kb + 3];
                au[4] = Erow[kb + 8]; au[5] = Erow[kb + 9];
                au[6] = Erow[kb + 10]; au[7] = Erow[kb + 11];
                // B fragment: col = lane%16, same packed k-pair layout as A
                int wb = ks * 16 + hi * 4;
                v8u bu;
                #pragma unroll
                for (int v = 0; v < 8; ++v) bu[v] = Wrow[wb + (v & 3) + (v >> 2) * 8];
                acc = __builtin_amdgcn_wmma_f32_16x16x32_bf16(
                    false, __builtin_bit_cast(v16bf, au),
                    false, __builtin_bit_cast(v16bf, bu),
                    (short)0, acc, false, false);
            }
            // D layout: row = r + 8*(lane>=16), col = lane%16
            int col = lane & 15, rb = hi * 8;
            #pragma unroll
            for (int r = 0; r < 8; ++r)
                stage[wave][(mt * 16 + rb + r) * 16 + col] = acc[r];
        }
        __syncthreads();

        // Fused epilogue: max/min over k, BN stats atomics
        for (int cell = lane; cell < 8 * 16; cell += 32) {
            int nl = cell >> 4, ol = cell & 15;
            float mx = -__builtin_inff(), mn = __builtin_inff(), s = 0.f, s2 = 0.f;
            #pragma unroll 4
            for (int kk = 0; kk < KNNK; ++kk) {
                float v = stage[wave][(nl * KNNK + kk) * 16 + ol];
                mx = fmaxf(mx, v); mn = fminf(mn, v); s += v; s2 += v * v;
            }
            size_t oidx = (size_t)(b * NN + n0 + nl) * O + obase + ol;
            outmax[oidx] = mx;
            outmin[oidx] = mn;
            atomicAdd(&stats[obase + ol], s);
            atomicAdd(&stats[1024 + obase + ol], s2);
        }
        __syncthreads();
    }
}

// ---------------------------------------------------------------------------
// BN affine + LeakyReLU applied to max-over-k (or min if scale < 0)
// ---------------------------------------------------------------------------
__global__ void bn_finalize(const float* __restrict__ outmax, const float* __restrict__ outmin,
                            const float* __restrict__ stats,
                            const float* __restrict__ gamma, const float* __restrict__ beta,
                            int O, float cnt,
                            float* __restrict__ xf, unsigned short* __restrict__ xbf, int ld) {
    int i = blockIdx.x * blockDim.x + threadIdx.x;
    if (i >= BB * NN * O) return;
    int o = i % O; size_t row = (size_t)i / O;
    float mean = stats[o] / cnt;
    float var  = stats[1024 + o] / cnt - mean * mean;
    float s = gamma[o] * rsqrtf(var + BNEPS);
    float v = (s >= 0.f) ? outmax[i] : outmin[i];
    float hn = (v - mean) * s + beta[o];
    hn = (hn >= 0.f) ? hn : SLOPE * hn;
    xf[row * ld + o]  = hn;
    xbf[row * ld + o] = f2bf(hn);
}

// ---------------------------------------------------------------------------
// Final 512->1024 GEMM via WMMA, h written to ws + BN stats atomics
// ---------------------------------------------------------------------------
__global__ __launch_bounds__(128) void final_gemm(const unsigned short* __restrict__ xcatb,
                                                  const unsigned short* __restrict__ W5b,
                                                  float* __restrict__ h5, float* __restrict__ stats) {
    int wave = threadIdx.x >> 5, lane = threadIdx.x & 31;
    int mt = blockIdx.x;                        // 2048 M-tiles of 16 rows
    int obase = blockIdx.y * 256 + wave * 64;   // 64 cols per wave = 4 O-tiles
    int m = lane & 15, hi = lane >> 4;

    const unsigned* Arow = (const unsigned*)(xcatb + (size_t)(mt * 16 + m) * 512);
    v8f acc[4];
    #pragma unroll
    for (int t = 0; t < 4; ++t) acc[t] = v8f{0.f, 0.f, 0.f, 0.f, 0.f, 0.f, 0.f, 0.f};

    for (int ks = 0; ks < 16; ++ks) {
        int kb = ks * 16 + hi * 4;
        v8u au;
        au[0] = Arow[kb + 0]; au[1] = Arow[kb + 1];
        au[2] = Arow[kb + 2]; au[3] = Arow[kb + 3];
        au[4] = Arow[kb + 8]; au[5] = Arow[kb + 9];
        au[6] = Arow[kb + 10]; au[7] = Arow[kb + 11];
        v16bf a = __builtin_bit_cast(v16bf, au);
        int wb = ks * 16 + hi * 4;
        #pragma unroll
        for (int t = 0; t < 4; ++t) {
            const unsigned* Wrow = (const unsigned*)(W5b + (size_t)(obase + t * 16 + m) * 512);
            v8u bu;
            #pragma unroll
            for (int v = 0; v < 8; ++v) bu[v] = Wrow[wb + (v & 3) + (v >> 2) * 8];
            acc[t] = __builtin_amdgcn_wmma_f32_16x16x32_bf16(
                false, a, false, __builtin_bit_cast(v16bf, bu), (short)0, acc[t], false, false);
        }
    }

    int col = lane & 15, rb = hi * 8;
    #pragma unroll
    for (int t = 0; t < 4; ++t) {
        int o = obase + t * 16 + col;
        float s = 0.f, s2 = 0.f;
        #pragma unroll
        for (int r = 0; r < 8; ++r) {
            float v = acc[t][r];
            h5[(size_t)(mt * 16 + rb + r) * 1024 + o] = v;
            s += v; s2 += v * v;
        }
        s  += __shfl_xor(s, 16);
        s2 += __shfl_xor(s2, 16);
        if (lane < 16) {
            atomicAdd(&stats[o], s);
            atomicAdd(&stats[1024 + o], s2);
        }
    }
}

// ---------------------------------------------------------------------------
// Final BN + LeakyReLU + max over the 1024 points -> d_out (B,1,1024)
// ---------------------------------------------------------------------------
__global__ void final_reduce(const float* __restrict__ h5, const float* __restrict__ stats,
                             const float* __restrict__ g5, const float* __restrict__ b5,
                             float* __restrict__ out) {
    int i = blockIdx.x * blockDim.x + threadIdx.x;
    if (i >= BB * 1024) return;
    int b = i / 1024, o = i % 1024;
    float cnt = (float)(BB * NN);
    float mean = stats[o] / cnt;
    float var  = stats[1024 + o] / cnt - mean * mean;
    float s = g5[o] * rsqrtf(var + BNEPS);
    float bt = b5[o];
    float best = -__builtin_inff();
    for (int n = 0; n < NN; ++n) {
        float v = h5[((size_t)b * NN + n) * 1024 + o];
        float hn = (v - mean) * s + bt;
        hn = (hn >= 0.f) ? hn : SLOPE * hn;
        best = fmaxf(best, hn);
    }
    out[(size_t)b * 1024 + o] = best;
}

// ---------------------------------------------------------------------------
static inline int cdiv(int a, int b) { return (a + b - 1) / b; }

extern "C" void kernel_launch(void* const* d_in, const int* in_sizes, int n_in,
                              void* d_out, int out_size, void* d_ws, size_t ws_size,
                              hipStream_t stream) {
    (void)in_sizes; (void)n_in; (void)out_size; (void)ws_size;
    const float* x  = (const float*)d_in[0];
    const float* W1 = (const float*)d_in[1],  *g1 = (const float*)d_in[2],  *b1 = (const float*)d_in[3];
    const float* W2 = (const float*)d_in[4],  *g2 = (const float*)d_in[5],  *b2 = (const float*)d_in[6];
    const float* W3 = (const float*)d_in[7],  *g3 = (const float*)d_in[8],  *b3 = (const float*)d_in[9];
    const float* W4 = (const float*)d_in[10], *g4 = (const float*)d_in[11], *b4 = (const float*)d_in[12];
    const float* W5 = (const float*)d_in[13], *g5 = (const float*)d_in[14], *b5 = (const float*)d_in[15];

    uint8_t* base = (uint8_t*)d_ws;
    size_t off = 0;
    auto alloc = [&](size_t bytes) -> uint8_t* {
        uint8_t* p = base + off;
        off = (off + bytes + 255) & ~(size_t)255;
        return p;
    };
    unsigned short* xb0   = (unsigned short*)alloc((size_t)BB * NN * 16 * 2);
    unsigned short* xcatb = (unsigned short*)alloc((size_t)BB * NN * 512 * 2);
    float*          xcatf = (float*)        alloc((size_t)BB * NN * 512 * 4);
    int*            idxb  = (int*)          alloc((size_t)BB * NN * KNNK * 4);
    unsigned short* W1b   = (unsigned short*)alloc((size_t)64 * 32 * 2);
    unsigned short* W2b   = (unsigned short*)alloc((size_t)64 * 128 * 2);
    unsigned short* W3b   = (unsigned short*)alloc((size_t)128 * 128 * 2);
    unsigned short* W4b   = (unsigned short*)alloc((size_t)256 * 256 * 2);
    unsigned short* W5b   = (unsigned short*)alloc((size_t)1024 * 512 * 2);
    float*          stats = (float*)        alloc((size_t)2048 * 4);
    float*          h5    = (float*)        alloc((size_t)BB * NN * 1024 * 4);
    float*          omax  = h5;                                   // dead by final phase
    float*          omin  = h5 + (size_t)BB * NN * 256;

    const int TB = 256;
    const float ECNT = (float)(BB * NN * KNNK);

    pack_rows<<<cdiv(BB * NN * 16, TB), TB, 0, stream>>>(x, xb0, BB * NN, 3, 16);
    pack_w_split<<<cdiv(64 * 32, TB), TB, 0, stream>>>(W1, W1b, 64, 3, 16);
    pack_w_split<<<cdiv(64 * 128, TB), TB, 0, stream>>>(W2, W2b, 64, 64, 64);
    pack_w_split<<<cdiv(128 * 128, TB), TB, 0, stream>>>(W3, W3b, 128, 64, 64);
    pack_w_split<<<cdiv(256 * 256, TB), TB, 0, stream>>>(W4, W4b, 256, 128, 128);
    pack_rows<<<cdiv(1024 * 512, TB), TB, 0, stream>>>(W5, W5b, 1024, 512, 512);

    const int KG = BB * NN / 8;

    // ---- Layer 1: C=3 (pad 16), O=64 ----
    knn_kernel<<<KG, 256, 0, stream>>>(x, 3, 3, idxb);
    hipMemsetAsync(stats, 0, 2048 * 4, stream);
    edge_gemm<16><<<KG, 128, 0, stream>>>(xb0, 16, idxb, W1b, 64, omax, omin, stats);
    bn_finalize<<<cdiv(BB * NN * 64, TB), TB, 0, stream>>>(omax, omin, stats, g1, b1, 64, ECNT,
                                                           xcatf + 0, xcatb + 0, 512);
    // ---- Layer 2: C=64, O=64 ----
    knn_kernel<<<KG, 256, 0, stream>>>(xcatf + 0, 64, 512, idxb);
    hipMemsetAsync(stats, 0, 2048 * 4, stream);
    edge_gemm<64><<<KG, 128, 0, stream>>>(xcatb + 0, 512, idxb, W2b, 64, omax, omin, stats);
    bn_finalize<<<cdiv(BB * NN * 64, TB), TB, 0, stream>>>(omax, omin, stats, g2, b2, 64, ECNT,
                                                           xcatf + 64, xcatb + 64, 512);
    // ---- Layer 3: C=64, O=128 ----
    knn_kernel<<<KG, 256, 0, stream>>>(xcatf + 64, 64, 512, idxb);
    hipMemsetAsync(stats, 0, 2048 * 4, stream);
    edge_gemm<64><<<KG, 128, 0, stream>>>(xcatb + 64, 512, idxb, W3b, 128, omax, omin, stats);
    bn_finalize<<<cdiv(BB * NN * 128, TB), TB, 0, stream>>>(omax, omin, stats, g3, b3, 128, ECNT,
                                                            xcatf + 128, xcatb + 128, 512);
    // ---- Layer 4: C=128, O=256 ----
    knn_kernel<<<KG, 256, 0, stream>>>(xcatf + 128, 128, 512, idxb);
    hipMemsetAsync(stats, 0, 2048 * 4, stream);
    edge_gemm<128><<<KG, 128, 0, stream>>>(xcatb + 128, 512, idxb, W4b, 256, omax, omin, stats);
    bn_finalize<<<cdiv(BB * NN * 256, TB), TB, 0, stream>>>(omax, omin, stats, g4, b4, 256, ECNT,
                                                            xcatf + 256, xcatb + 256, 512);
    // ---- Final linear 512->1024 + BN + max over points ----
    hipMemsetAsync(stats, 0, 2048 * 4, stream);
    final_gemm<<<dim3(BB * NN / 16, 4), 128, 0, stream>>>(xcatb, W5b, h5, stats);
    final_reduce<<<cdiv(BB * 1024, TB), TB, 0, stream>>>(h5, stats, g5, b5, (float*)d_out);
}
